// GatedAttentionQwenStyle_16870631539494
// MI455X (gfx1250) — matrix operand order repaired and checked by reading, not compiled, not run on hardware
//
#include <hip/hip_runtime.h>
#include <hip/hip_bf16.h>

// ---------------------------------------------------------------------------
// Gated attention (Qwen-style) on gfx1250, bf16 WMMA everywhere.
// B=2 S=2048 HID=2048 NH=16 NKV=8 HD=128 GATE=16 QKV_OUT=4112
// Pipeline: convert -> QKV GEMM (WMMA, TDM tensor_load_to_lds double-buffered)
//           -> RoPE -> flash attention (WMMA, per-lane async-LDS K staging,
//           online softmax, fused sigmoid gate) -> out GEMM (WMMA).
// ---------------------------------------------------------------------------

typedef __bf16 bf16;
typedef __bf16 v8bf  __attribute__((ext_vector_type(8)));
typedef __bf16 v16bf __attribute__((ext_vector_type(16)));
typedef float  v8f   __attribute__((ext_vector_type(8)));
typedef unsigned int u32x4 __attribute__((ext_vector_type(4)));
typedef int          i32x4 __attribute__((ext_vector_type(4)));
typedef int          i32x8 __attribute__((ext_vector_type(8)));

constexpr int B_    = 2;
constexpr int S_    = 2048;
constexpr int HID_  = 2048;
constexpr int NH_   = 16;
constexpr int NKV_  = 8;
constexpr int HD_   = 128;
constexpr int GATE_ = 16;
constexpr int KV_DIM_  = NKV_ * HD_;                       // 1024
constexpr int QKV_OUT_ = NH_ * HD_ + GATE_ + 2 * KV_DIM_;  // 4112
constexpr int M_       = B_ * S_;                          // 4096

// ---------------------------------------------------------------------------
// CDNA5 helpers
// ---------------------------------------------------------------------------
static __device__ inline v8f wmma_bf16(v16bf a, v16bf b, v8f c) {
  return __builtin_amdgcn_wmma_f32_16x16x32_bf16(false, a, false, b,
                                                 (short)0, c, false, false);
}

static __device__ inline v16bf cat8(v8bf lo, v8bf hi) {
  return __builtin_shufflevector(lo, hi, 0, 1, 2, 3, 4, 5, 6, 7,
                                 8, 9, 10, 11, 12, 13, 14, 15);
}

// Generic flat address truncated to 32b == LDS byte offset (ISA 10.2: LDS
// aperture lives in addr[63:32]; hardware adds LDS_BASE).
static __device__ inline uint32_t lds_addr(const void* p) {
  return (uint32_t)(uintptr_t)p;
}

// Per-lane async DMA: 16B per lane, memory -> LDS, tracked by ASYNCcnt.
static __device__ inline void async_copy_b128(uint32_t lds_off, const bf16* g) {
  asm volatile("global_load_async_to_lds_b128 %0, %1, off"
               :: "v"(lds_off), "v"((unsigned long long)(uintptr_t)g)
               : "memory");
}

// Tensor Data Mover: one descriptor DMAs a 2D tile (row_bytes x rows_tile,
// source row pitch stride_bytes) into LDS, inserting 16B of padding after
// every 64B of data (pad_interval=3 -> 16 DWORDs, pad_amount=3 -> 4 DWORDs)
// to reproduce the bank-padded LDST=40 row layout. Rows beyond rows_tensor
// read as zero (OOB semantics) -> exact N-edge handling.
// Tracked by TENSORcnt; issue from a single wave.
static __device__ inline void tdm_load_2d_pad(uint32_t lds_off, const void* gaddr,
                                              uint32_t row_bytes, uint32_t rows_tile,
                                              uint32_t rows_tensor,
                                              uint32_t stride_bytes) {
  const uint64_t ga = (uint64_t)(uintptr_t)gaddr;
  u32x4 g0;
  g0[0] = 1u;                                               // count=1 (valid D#)
  g0[1] = lds_off;                                          // lds_addr
  g0[2] = (uint32_t)ga;                                     // global_addr[31:0]
  g0[3] = (uint32_t)((ga >> 32) & 0x01FFFFFFu) | (2u << 30);  // addr[56:32]|type=2
  i32x8 g1;
  g1[0] = (int)((1u << 20) | (3u << 22) | (3u << 25));      // pad en/interval/amount
  g1[1] = (int)((row_bytes & 0xFFFFu) << 16);               // tensor_dim0[15:0]
  g1[2] = (int)((row_bytes >> 16) | ((rows_tensor & 0xFFFFu) << 16));  // td0 hi|td1 lo
  g1[3] = (int)(((rows_tensor >> 16) & 0xFFFFu) | ((row_bytes & 0xFFFFu) << 16));
                                                            // td1 hi | tile_dim0
  g1[4] = (int)(rows_tile & 0xFFFFu);                       // tile_dim1 (tile_dim2=0)
  g1[5] = (int)stride_bytes;                                // tensor_dim0_stride lo
  g1[6] = 0;                                                // stride hi | td1_stride lo
  g1[7] = 0;
  i32x4 z4 = {0, 0, 0, 0};
#if defined(__clang_major__) && (__clang_major__ >= 23)
  i32x8 z8 = {0, 0, 0, 0, 0, 0, 0, 0};
  __builtin_amdgcn_tensor_load_to_lds(g0, g1, z4, z4, z8, 0);
#else
  __builtin_amdgcn_tensor_load_to_lds(g0, g1, z4, z4, 0);
#endif
}

// A-fragment (16x32 MxK, 16-bit): lane&15 = M row.
// elem e: K = 8*(lane>=16) + (e&7) + 16*(e>=8)  -> two contiguous 16B chunks.
static __device__ inline v16bf load_a_frag(const bf16* base, int stride, int lane) {
  const int lm = lane & 15, hs = lane >> 4;
  const bf16* r = base + lm * stride + hs * 8;
  return cat8(*(const v8bf*)(r), *(const v8bf*)(r + 16));
}

// B-fragment (32x16 KxN, 16-bit): lane&15 = N row (stored N-major, K contig).
// elem e: K = 8*(lane>=16) + (e&7) + 16*(e>=8)  -> two contiguous 16B chunks.
static __device__ inline v16bf load_b_frag(const bf16* base, int stride, int lane) {
  const int lm = lane & 15, hs = lane >> 4;
  const bf16* r = base + lm * stride + hs * 8;
  return cat8(*(const v8bf*)(r), *(const v8bf*)(r + 16));
}

static __device__ inline v8f v8f_zero() {
  v8f z = {0.f, 0.f, 0.f, 0.f, 0.f, 0.f, 0.f, 0.f};
  return z;
}

// ---------------------------------------------------------------------------
// fp32 -> bf16 convert
// ---------------------------------------------------------------------------
__global__ __launch_bounds__(256) void f32_to_bf16_kernel(const float* __restrict__ in,
                                                          bf16* __restrict__ out, int n) {
  int i = blockIdx.x * 256 + threadIdx.x;
  if (i < n) out[i] = (bf16)in[i];
}

// ---------------------------------------------------------------------------
// bf16 WMMA GEMM: C[M,N] = A[M,K] @ Bw[N,K]^T
// Block tile 128x128, K-step 32, 256 threads = 8 waves, wave tile 64x32.
// LDS double-buffered, fed by one TDM descriptor per operand tile (issued by
// wave 0; s_wait_tensorcnt 2 retires tile t while t+1 streams). TDM LDS
// padding reproduces the LDST=40 layout; TDM OOB zero-fill handles the N edge.
// ---------------------------------------------------------------------------
constexpr int LDST = 40;  // padded LDS stride (bf16), 80B rows -> 16B aligned

template <bool OUT_BF16>
__global__ __launch_bounds__(256) void gemm_bf16_wmma(const bf16* __restrict__ A,
                                                      const bf16* __restrict__ Bw,
                                                      void* __restrict__ Cout,
                                                      int M, int N, int K) {
  __shared__ bf16 As[2][128 * LDST];
  __shared__ bf16 Bs[2][128 * LDST];

  const int tid  = threadIdx.x;
  const int lane = tid & 31;
  const int wave = tid >> 5;
  const int wm   = (wave >> 2) * 64;  // 0 / 64
  const int wn   = (wave & 3) * 32;   // 0..96
  const int bm   = blockIdx.y * 128;
  const int bn   = blockIdx.x * 128;
  const uint32_t rowsB = (uint32_t)((N - bn < 128) ? (N - bn) : 128);

  v8f acc[4][2];
#pragma unroll
  for (int mf = 0; mf < 4; ++mf)
#pragma unroll
    for (int nf = 0; nf < 2; ++nf) acc[mf][nf] = v8f_zero();

  auto stage = [&](int k0, int buf) {
    // A: 128 rows x 64B, pitch K*2; full rows valid.
    tdm_load_2d_pad(lds_addr(&As[buf][0]), A + (size_t)bm * K + k0,
                    64u, 128u, 128u, (uint32_t)K * 2u);
    // B: rows beyond N-bn zero-filled by TDM OOB handling.
    tdm_load_2d_pad(lds_addr(&Bs[buf][0]), Bw + (size_t)bn * K + k0,
                    64u, 128u, rowsB, (uint32_t)K * 2u);
  };

  const int T = K / 32;
  if (wave == 0) stage(0, 0);
  for (int t = 0; t < T; ++t) {
    const int cur = t & 1;
    if (wave == 0) {
      if (t + 1 < T) {
        stage((t + 1) * 32, cur ^ 1);
        __builtin_amdgcn_s_wait_tensorcnt(2);  // tile t's 2 descriptors landed
      } else {
        __builtin_amdgcn_s_wait_tensorcnt(0);
      }
    }
    __syncthreads();

    v16bf af[4];
#pragma unroll
    for (int mf = 0; mf < 4; ++mf)
      af[mf] = load_a_frag(&As[cur][(wm + mf * 16) * LDST], LDST, lane);
    v16bf bfr[2];
#pragma unroll
    for (int nf = 0; nf < 2; ++nf)
      bfr[nf] = load_b_frag(&Bs[cur][(wn + nf * 16) * LDST], LDST, lane);
#pragma unroll
    for (int mf = 0; mf < 4; ++mf)
#pragma unroll
      for (int nf = 0; nf < 2; ++nf)
        acc[mf][nf] = wmma_bf16(af[mf], bfr[nf], acc[mf][nf]);
    __syncthreads();
  }

  // C-fragment: elem i -> row = i + 8*(lane>=16), col = lane&15
  const int lm = lane & 15, hs = lane >> 4;
#pragma unroll
  for (int mf = 0; mf < 4; ++mf)
#pragma unroll
    for (int nf = 0; nf < 2; ++nf)
#pragma unroll
      for (int i = 0; i < 8; ++i) {
        int row = bm + wm + mf * 16 + hs * 8 + i;
        int col = bn + wn + nf * 16 + lm;
        if (col < N) {
          if (OUT_BF16)
            ((bf16*)Cout)[(size_t)row * N + col] = (bf16)acc[mf][nf][i];
          else
            ((float*)Cout)[(size_t)row * N + col] = acc[mf][nf][i];
        }
      }
}

// ---------------------------------------------------------------------------
// RoPE + head split: qkv[M, QKV_OUT] -> q_r[B,NH,S,HD], k_r[B,NKV,S,HD],
// v_h[B,NKV,S,HD] (head-major bf16 for attention).
// ---------------------------------------------------------------------------
__global__ __launch_bounds__(256) void rope_split_kernel(const bf16* __restrict__ qkv,
                                                         const float* __restrict__ cosb,
                                                         const float* __restrict__ sinb,
                                                         bf16* __restrict__ q_r,
                                                         bf16* __restrict__ k_r,
                                                         bf16* __restrict__ v_h) {
  int tid = blockIdx.x * 256 + threadIdx.x;
  int d   = tid & (HD_ - 1);
  int h32 = (tid >> 7) & 31;
  int m   = tid >> 12;  // HD_*32 = 4096
  if (m >= M_) return;
  int b = m / S_, s = m % S_;
  const bf16* row = qkv + (size_t)m * QKV_OUT_;

  if (h32 < NH_ + NKV_) {  // RoPE for q and k
    int base = (h32 < NH_) ? h32 * HD_ : HID_ + GATE_ + (h32 - NH_) * HD_;
    float x  = (float)row[base + d];
    int   d2 = (d < HD_ / 2) ? d + HD_ / 2 : d - HD_ / 2;
    float xr = (float)row[base + d2];
    float rot = (d < HD_ / 2) ? -xr : xr;
    float val = x * cosb[s * HD_ + d] + rot * sinb[s * HD_ + d];
    if (h32 < NH_)
      q_r[(((size_t)b * NH_ + h32) * S_ + s) * HD_ + d] = (bf16)val;
    else
      k_r[(((size_t)b * NKV_ + (h32 - NH_)) * S_ + s) * HD_ + d] = (bf16)val;
  } else {  // v passthrough
    int hh = h32 - NH_ - NKV_;
    int base = HID_ + GATE_ + KV_DIM_ + hh * HD_;
    v_h[(((size_t)b * NKV_ + hh) * S_ + s) * HD_ + d] = row[base + d];
  }
}

// ---------------------------------------------------------------------------
// Flash attention with WMMA + fused sigmoid gate.
// Grid: (S/128, NH, B), 256 threads = 8 waves; each wave owns 16 q rows x HD.
// K tiles stream via per-lane async-to-LDS double buffer; V staged transposed
// with packed b32 stores; P converted C-layout -> A-fragment via LDS.
// ---------------------------------------------------------------------------
__global__ __launch_bounds__(256) void attention_kernel(const bf16* __restrict__ q_r,
                                                        const bf16* __restrict__ k_r,
                                                        const bf16* __restrict__ v_h,
                                                        const bf16* __restrict__ qkv,
                                                        bf16* __restrict__ attn_out) {
  __shared__ bf16 Ks[2][32 * 136];    // [key 32][hd 128] (+8 pad), async-fed
  __shared__ bf16 Vs[128 * LDST];     // [hd 128][key 32] (+8 pad), transposed
  __shared__ bf16 Ps[8 * 16 * LDST];  // per-wave 16x32 P tile

  const int b    = blockIdx.z;
  const int h    = blockIdx.y;
  const int qblk = blockIdx.x * 128;
  const int tid  = threadIdx.x;
  const int lane = tid & 31;
  const int wave = tid >> 5;
  const int lm   = lane & 15, hs = lane >> 4;
  const int hk   = h >> 1;  // GQA: NH/NKV = 2

  const bf16* qbase = q_r + ((size_t)(b * NH_ + h) * S_ + qblk + wave * 16) * HD_;
  const bf16* kbase = k_r + ((size_t)(b * NKV_ + hk) * S_) * HD_;
  const bf16* vbase = v_h + ((size_t)(b * NKV_ + hk) * S_) * HD_;

  // Q A-fragments: 16 rows x 128 HD = 4 chunks of K=32
  v16bf qf[4];
#pragma unroll
  for (int c = 0; c < 4; ++c) qf[c] = load_a_frag(qbase + c * 32, HD_, lane);

  v8f o[8];
#pragma unroll
  for (int f = 0; f < 8; ++f) o[f] = v8f_zero();
  float m_run[8], l_run[8];
#pragma unroll
  for (int i = 0; i < 8; ++i) { m_run[i] = -3.0e38f; l_run[i] = 0.f; }

  bf16* Pw = Ps + wave * 16 * LDST;

  const int ldr = tid >> 3;        // 0..31 key row (K staging)
  const int ldc = (tid & 7) * 16;  // hd chunk (K staging)
  const int kp  = tid & 15;        // key pair (V staging)
  const int hc  = (tid >> 4) * 8;  // hd chunk of 8 (V staging)
  const float scale = 0.08838834764831845f;  // 1/sqrt(128)

  auto stage_k = [&](int kt, int buf) {
    const bf16* gk = kbase + (size_t)(kt + ldr) * HD_ + ldc;
    uint32_t lk = lds_addr(&Ks[buf][ldr * 136 + ldc]);
    async_copy_b128(lk, gk);
    async_copy_b128(lk + 16, gk + 8);
  };

  stage_k(0, 0);
  for (int kt = 0; kt < S_; kt += 32) {
    const int cur = (kt >> 5) & 1;
    __syncthreads();  // everyone done with Ks[cur^1] and Vs from last tile
    if (kt + 32 < S_) stage_k(kt + 32, cur ^ 1);

    {  // V: load 2 adjacent key rows, store transposed as packed b32
      const bf16* gv0 = vbase + (size_t)(kt + 2 * kp) * HD_ + hc;
      v8bf a0 = *(const v8bf*)(gv0);
      v8bf a1 = *(const v8bf*)(gv0 + HD_);
#pragma unroll
      for (int j = 0; j < 8; ++j) {
        union { bf16 h2[2]; uint32_t u; } pk2;
        pk2.h2[0] = a0[j];
        pk2.h2[1] = a1[j];
        *(uint32_t*)&Vs[(hc + j) * LDST + 2 * kp] = pk2.u;
      }
    }
    if (kt + 32 < S_)
      asm volatile("s_wait_asynccnt 0x2" ::: "memory");  // tile kt's K landed
    else
      asm volatile("s_wait_asynccnt 0x0" ::: "memory");
    __syncthreads();

    // scores: two 16x16 tiles (32 keys), K-loop over HD
    v8f s0 = v8f_zero(), s1 = v8f_zero();
#pragma unroll
    for (int c = 0; c < 4; ++c) {
      v16bf kb0 = load_b_frag(&Ks[cur][0 * 16 * 136 + c * 32], 136, lane);
      v16bf kb1 = load_b_frag(&Ks[cur][1 * 16 * 136 + c * 32], 136, lane);
      s0 = wmma_bf16(qf[c], kb0, s0);
      s1 = wmma_bf16(qf[c], kb1, s1);
    }

    // online softmax (rows live in 16-lane halves; masks 1/2/4/8 stay inside)
    float p0[8], p1[8], corr[8];
#pragma unroll
    for (int i = 0; i < 8; ++i) {
      float a0 = s0[i] * scale, a1 = s1[i] * scale;
      float r = fmaxf(a0, a1);
      r = fmaxf(r, __shfl_xor(r, 1));
      r = fmaxf(r, __shfl_xor(r, 2));
      r = fmaxf(r, __shfl_xor(r, 4));
      r = fmaxf(r, __shfl_xor(r, 8));
      float mn = fmaxf(m_run[i], r);
      corr[i]  = __expf(m_run[i] - mn);
      m_run[i] = mn;
      p0[i] = __expf(a0 - mn);
      p1[i] = __expf(a1 - mn);
      float rs = p0[i] + p1[i];
      rs += __shfl_xor(rs, 1);
      rs += __shfl_xor(rs, 2);
      rs += __shfl_xor(rs, 4);
      rs += __shfl_xor(rs, 8);
      l_run[i] = l_run[i] * corr[i] + rs;
    }
#pragma unroll
    for (int f = 0; f < 8; ++f)
#pragma unroll
      for (int i = 0; i < 8; ++i) o[f][i] *= corr[i];

    // C-layout -> A-fragment via wave-private LDS tile
#pragma unroll
    for (int i = 0; i < 8; ++i) {
      int r = hs * 8 + i;
      Pw[r * LDST + lm]      = (bf16)p0[i];
      Pw[r * LDST + 16 + lm] = (bf16)p1[i];
    }
    asm volatile("s_wait_dscnt 0x0" ::: "memory");  // wave-local RAW on LDS
    v16bf pf = load_a_frag(Pw, LDST, lane);

    // PV: 8 HD tiles of 16 cols, K = 32 keys
#pragma unroll
    for (int f = 0; f < 8; ++f) {
      v16bf vf = load_b_frag(&Vs[f * 16 * LDST], LDST, lane);
      o[f] = wmma_bf16(pf, vf, o[f]);
    }
  }

  // epilogue: 1/l normalization fused with sigmoid(gate)
  float g[8];
#pragma unroll
  for (int i = 0; i < 8; ++i) {
    int s_tok  = qblk + wave * 16 + hs * 8 + i;
    float gate = (float)qkv[(size_t)(b * S_ + s_tok) * QKV_OUT_ + HID_ + h];
    g[i] = (1.f / (1.f + __expf(-gate))) / l_run[i];
  }
#pragma unroll
  for (int f = 0; f < 8; ++f)
#pragma unroll
    for (int i = 0; i < 8; ++i) {
      int s_tok = qblk + wave * 16 + hs * 8 + i;
      attn_out[(size_t)(b * S_ + s_tok) * (NH_ * HD_) + h * HD_ + f * 16 + lm] =
          (bf16)(o[f][i] * g[i]);
    }
}

// ---------------------------------------------------------------------------
// Host launcher
// ---------------------------------------------------------------------------
extern "C" void kernel_launch(void* const* d_in, const int* in_sizes, int n_in,
                              void* d_out, int out_size, void* d_ws, size_t ws_size,
                              hipStream_t stream) {
  (void)in_sizes; (void)n_in; (void)out_size; (void)ws_size;
  const float* hidden = (const float*)d_in[0];
  const float* cosb   = (const float*)d_in[1];
  const float* sinb   = (const float*)d_in[2];
  const float* w_qkv  = (const float*)d_in[3];
  const float* w_o    = (const float*)d_in[4];

  char*  ws  = (char*)d_ws;
  size_t off = 0;
  auto carve = [&](size_t bytes) -> char* {
    char* p = ws + off;
    off += (bytes + 255) & ~(size_t)255;
    return p;
  };
  bf16* hid_bf  = (bf16*)carve((size_t)M_ * HID_ * 2);
  bf16* wqkv_bf = (bf16*)carve((size_t)QKV_OUT_ * HID_ * 2);
  bf16* wo_bf   = (bf16*)carve((size_t)HID_ * NH_ * HD_ * 2);
  bf16* qkv_bf  = (bf16*)carve((size_t)M_ * QKV_OUT_ * 2);
  bf16* q_rope  = (bf16*)carve((size_t)B_ * NH_ * S_ * HD_ * 2);
  bf16* k_rope  = (bf16*)carve((size_t)B_ * NKV_ * S_ * HD_ * 2);
  bf16* v_head  = (bf16*)carve((size_t)B_ * NKV_ * S_ * HD_ * 2);
  bf16* attn_bf = (bf16*)carve((size_t)M_ * NH_ * HD_ * 2);

  const int n_hid = M_ * HID_;
  f32_to_bf16_kernel<<<(n_hid + 255) / 256, 256, 0, stream>>>(hidden, hid_bf, n_hid);
  const int n_wq = QKV_OUT_ * HID_;
  f32_to_bf16_kernel<<<(n_wq + 255) / 256, 256, 0, stream>>>(w_qkv, wqkv_bf, n_wq);
  const int n_wo = HID_ * NH_ * HD_;
  f32_to_bf16_kernel<<<(n_wo + 255) / 256, 256, 0, stream>>>(w_o, wo_bf, n_wo);

  // qkv = hidden @ w_qkv^T  (M=4096, N=4112, K=2048)
  gemm_bf16_wmma<true><<<dim3((QKV_OUT_ + 127) / 128, M_ / 128), 256, 0, stream>>>(
      hid_bf, wqkv_bf, qkv_bf, M_, QKV_OUT_, HID_);

  const int n_rope = B_ * S_ * 32 * HD_;
  rope_split_kernel<<<n_rope / 256, 256, 0, stream>>>(qkv_bf, cosb, sinb,
                                                      q_rope, k_rope, v_head);

  attention_kernel<<<dim3(S_ / 128, NH_, B_), 256, 0, stream>>>(
      q_rope, k_rope, v_head, qkv_bf, attn_bf);

  // out = gated_attn @ w_o^T  (M=4096, N=2048, K=2048), fp32 out
  gemm_bf16_wmma<false><<<dim3(HID_ / 128, M_ / 128), 256, 0, stream>>>(
      attn_bf, wo_bf, d_out, M_, HID_, NH_ * HD_);
}
